// EnhancedBrain_90314572300899
// MI455X (gfx1250) — matrix operand order repaired and checked by reading, not compiled, not run on hardware
//
#include <hip/hip_runtime.h>
#include <hip/hip_bf16.h>
#include <cstdint>
#include <cstddef>

// ---------------- problem constants (from reference) ----------------
#define D_MODEL   1024
#define D_FF      4096
#define N_EXPERTS 4
#define HIDDEN    256
#define BATCH     8
#define SEQ       512
#define NTOK      (BATCH * SEQ)   // 4096 tokens

typedef __attribute__((ext_vector_type(16))) __bf16 v16bf;
typedef __attribute__((ext_vector_type(8)))  float  v8f;

// CDNA5 async global->LDS path (guarded: falls back to register-staged pipeline)
#if defined(__has_builtin)
#if __has_builtin(__builtin_amdgcn_global_load_async_to_lds_b128)
#define ZB_ASYNC_LDS 1
#endif
#endif
#ifndef ZB_ASYNC_LDS
#define ZB_ASYNC_LDS 0
#endif

// builtin parameter types (from hipcc diagnostic): generic v4i* src, LDS v4i* dst
typedef int zb_v4i __attribute__((vector_size(16)));
typedef zb_v4i __attribute__((address_space(3))) * zb_v4i_lds;

__device__ __forceinline__ void zb_wait_async0() {
#if defined(__has_builtin)
#if __has_builtin(__builtin_amdgcn_s_wait_asynccnt)
  __builtin_amdgcn_s_wait_asynccnt(0);
  return;
#else
  asm volatile("s_wait_asynccnt 0" ::: "memory");
  return;
#endif
#else
  asm volatile("s_wait_asynccnt 0" ::: "memory");
#endif
}

__device__ __forceinline__ unsigned short zb_f2bf(float f) {
  unsigned int u = __builtin_bit_cast(unsigned int, f);
  unsigned int r = u + 0x7FFFu + ((u >> 16) & 1u);
  return (unsigned short)(r >> 16);
}

// branch-free tanh-approx gelu (jax.nn.gelu default)
__device__ __forceinline__ float zb_gelu(float v) {
  const float u = 0.7978845608028654f * (v + 0.044715f * v * v * v);
  const float t = 1.0f - 2.0f / (1.0f + __expf(2.0f * u));   // tanh(u)
  return 0.5f * v * (1.0f + t);
}

// ---------------- fp32 -> bf16 (plain, row-major kept) ----------------
__global__ void zb_f32_to_bf16(const float* __restrict__ src,
                               unsigned short* __restrict__ dst, int n) {
  int i = blockIdx.x * blockDim.x + threadIdx.x;
  int stride = gridDim.x * blockDim.x;
  for (; i < n; i += stride) dst[i] = zb_f2bf(src[i]);
}

// ---------------- fp32 [K][N] -> bf16 [N][K] (tiled transpose) ----------------
__global__ __launch_bounds__(256)
void zb_transpose_to_bf16(const float* __restrict__ src,       // [K][N]
                          unsigned short* __restrict__ dst,    // [N][K]
                          int K, int N) {
  __shared__ float tile[32][33];
  const int k0 = blockIdx.y * 32;
  const int n0 = blockIdx.x * 32;
  const int tx = threadIdx.x & 31;
  const int ty = threadIdx.x >> 5;   // 0..7
#pragma unroll
  for (int dy = 0; dy < 32; dy += 8)
    tile[ty + dy][tx] = src[(size_t)(k0 + ty + dy) * N + (n0 + tx)];
  __syncthreads();
#pragma unroll
  for (int dy = 0; dy < 32; dy += 8)
    dst[(size_t)(n0 + ty + dy) * K + (k0 + tx)] = zb_f2bf(tile[tx][ty + dy]);
}

// ---------------- out = x (residual init) ----------------
__global__ void zb_copy_f32(const float* __restrict__ src,
                            float* __restrict__ dst, int n4) {
  int i = blockIdx.x * blockDim.x + threadIdx.x;
  int stride = gridDim.x * blockDim.x;
  const float4* s = (const float4*)src;
  float4* d = (float4*)dst;
  for (; i < n4; i += stride) d[i] = s[i];
}

// ---------------- router: mean-pool -> tanh MLP -> softmax -> top-3 ----------------
__global__ void zb_router(const float* __restrict__ x,
                          const float* __restrict__ rw1, const float* __restrict__ rb1,
                          const float* __restrict__ rw2, const float* __restrict__ rb2,
                          float* __restrict__ w_be) {
  __shared__ float pooled[D_MODEL];
  __shared__ float h[HIDDEN];
  __shared__ float logits[N_EXPERTS];
  const int b = blockIdx.x;
  const int t = threadIdx.x;

  for (int d = t; d < D_MODEL; d += blockDim.x) {
    float s = 0.f;
    const float* xp = x + (size_t)b * SEQ * D_MODEL + d;
    for (int si = 0; si < SEQ; ++si) s += xp[(size_t)si * D_MODEL];
    pooled[d] = s * (1.0f / (float)SEQ);
  }
  __syncthreads();

  if (t < HIDDEN) {
    float acc = rb1[t];
    for (int d = 0; d < D_MODEL; ++d) acc += pooled[d] * rw1[(size_t)d * HIDDEN + t];
    h[t] = tanhf(acc);
  }
  __syncthreads();

  if (t < N_EXPERTS) {
    float acc = rb2[t];
    for (int j = 0; j < HIDDEN; ++j) acc += h[j] * rw2[(size_t)j * N_EXPERTS + t];
    logits[t] = acc;
  }
  __syncthreads();

  if (t == 0) {
    float mx = logits[0];
    for (int e = 1; e < N_EXPERTS; ++e) mx = fmaxf(mx, logits[e]);
    float p[N_EXPERTS], sum = 0.f;
    for (int e = 0; e < N_EXPERTS; ++e) { p[e] = __expf(logits[e] - mx); sum += p[e]; }
    for (int e = 0; e < N_EXPERTS; ++e) p[e] /= sum;
    int imin = 0;
    for (int e = 1; e < N_EXPERTS; ++e) if (p[e] < p[imin]) imin = e;
    float s3 = 0.f;
    for (int e = 0; e < N_EXPERTS; ++e) if (e != imin) s3 += p[e];
    for (int e = 0; e < N_EXPERTS; ++e)
      w_be[b * N_EXPERTS + e] = (e == imin) ? 0.f : p[e] / s3;
  }
}

// ---------------- WMMA GEMM core ----------------
// C = A[M][K](bf16,row-major) @ Bt[N][K](bf16, pre-transposed) (+ bias)
// MODE 1: hid(bf16) = gelu(...) via LDS-shuffled b128 stores
// MODE 2: out(f32) += w_be[b,e] * (...) with batched RMW
// Block 128x128, K-step 64, double-buffered async LDS staging,
// 8 waves of 32x64 -> 16 WMMAs / wave / stage.
template <int MODE>
__global__ __launch_bounds__(256)
void zb_moe_gemm(const unsigned short* __restrict__ A,    // [M][K]
                 const unsigned short* __restrict__ Btg,  // [N][K]
                 const float* __restrict__ bias,          // [N]
                 void* __restrict__ Cv,
                 const float* __restrict__ w_be,
                 int M, int N, int K, int expert) {
  constexpr int BM = 128, BN = 128, BK = 64, RS = BK + 8;   // RS=72 shorts (144B rows)
  __shared__ unsigned short As[2][BM][RS];   // 18 KB per buffer (also reused by epilogue)
  __shared__ unsigned short Bs[2][BN][RS];   // 18 KB per buffer

  const int tid = threadIdx.x;
  const int block_m = blockIdx.y * BM;
  const int block_n = blockIdx.x * BN;

  // one batch per block-row (SEQ % BM == 0): top-k gate, skip dropped expert
  const int b = block_m / SEQ;
  const float wgt = w_be[b * N_EXPERTS + expert];
  if (wgt == 0.0f) return;

  const int wave  = tid >> 5;
  const int lane  = tid & 31;
  const int wm    = wave >> 1;        // 4 x 32 rows
  const int wn    = wave & 1;         // 2 x 64 cols
  const int lrow  = lane & 15;
  const int lhalf = lane >> 4;

  // staging: 2 threads per row, 32 cols (4 x uint4) each
  const int srow = tid >> 1;          // 0..127
  const int scol = (tid & 1) * 32;    // 0 or 32
  const unsigned short* aRow = A   + (size_t)(block_m + srow) * K + scol;
  const unsigned short* bRow = Btg + (size_t)(block_n + srow) * K + scol;

  v8f acc[2][4] = {};

#if ZB_ASYNC_LDS
  // ---- CDNA5 async global->LDS (ASYNCcnt) ----
  auto stage_async = [&](int buf, int k0) {
#pragma unroll
    for (int j = 0; j < 4; ++j) {
      __builtin_amdgcn_global_load_async_to_lds_b128(
          (zb_v4i*)(aRow + k0 + j * 8),
          (zb_v4i_lds)(&As[buf][srow][scol + j * 8]), 0, 0);
      __builtin_amdgcn_global_load_async_to_lds_b128(
          (zb_v4i*)(bRow + k0 + j * 8),
          (zb_v4i_lds)(&Bs[buf][srow][scol + j * 8]), 0, 0);
    }
  };
  stage_async(0, 0);
  zb_wait_async0();
  __syncthreads();
#else
  uint4 ra[4], rb[4];
  auto stage_load = [&](int k0) {
#pragma unroll
    for (int j = 0; j < 4; ++j) {
      ra[j] = *(const uint4*)(aRow + k0 + j * 8);
      rb[j] = *(const uint4*)(bRow + k0 + j * 8);
    }
  };
  auto stage_store = [&](int buf) {
#pragma unroll
    for (int j = 0; j < 4; ++j) {
      *(uint4*)(&As[buf][srow][scol + j * 8]) = ra[j];
      *(uint4*)(&Bs[buf][srow][scol + j * 8]) = rb[j];
    }
  };
  stage_load(0);
  stage_store(0);
  __syncthreads();
#endif

  const int nStages = K / BK;
  for (int s = 0; s < nStages; ++s) {
    const int cur = s & 1;

    // kick off next stage while we compute this one
    if (s + 1 < nStages) {
#if ZB_ASYNC_LDS
      stage_async(cur ^ 1, (s + 1) * BK);
#else
      stage_load((s + 1) * BK);
#endif
      if (s + 2 < nStages) {   // global_prefetch_b8 for stage s+2
        __builtin_prefetch(aRow + (s + 2) * BK, 0, 3);
        __builtin_prefetch(bRow + (s + 2) * BK, 0, 3);
      }
    }

    // ---- compute on buffer `cur`: 2 sub-steps of K=32 ----
#pragma unroll
    for (int kk = 0; kk < BK; kk += 32) {
      union FragU { v16bf v; uint4 q[2]; };
      FragU afrag[2], bfrag[4];
#pragma unroll
      for (int mi = 0; mi < 2; ++mi) {
        const int row = wm * 32 + mi * 16 + lrow;
        // lanes 0-15: K 0-7 / 16-23 ; lanes 16-31: K 8-15 / 24-31
        afrag[mi].q[0] = *(const uint4*)&As[cur][row][kk + lhalf * 8];
        afrag[mi].q[1] = *(const uint4*)&As[cur][row][kk + 16 + lhalf * 8];
      }
#pragma unroll
      for (int ni = 0; ni < 4; ++ni) {
        const int col = wn * 64 + ni * 16 + lrow;
        // lanes 0-15: K 0-7 / 16-23 ; lanes 16-31: K 8-15 / 24-31
        bfrag[ni].q[0] = *(const uint4*)&Bs[cur][col][kk + lhalf * 8];
        bfrag[ni].q[1] = *(const uint4*)&Bs[cur][col][kk + 16 + lhalf * 8];
      }
#pragma unroll
      for (int mi = 0; mi < 2; ++mi)
#pragma unroll
        for (int ni = 0; ni < 4; ++ni)
          acc[mi][ni] = __builtin_amdgcn_wmma_f32_16x16x32_bf16(
              false, afrag[mi].v, false, bfrag[ni].v,
              (short)0, acc[mi][ni], false, false);
    }

    // ---- publish next buffer ----
    if (s + 1 < nStages) {
#if ZB_ASYNC_LDS
      zb_wait_async0();
      __syncthreads();
#else
      stage_store(cur ^ 1);
      __syncthreads();
#endif
    }
  }

  // ---- epilogue (C/D layout: elem i, lane l -> M = 8*(l/16)+i, N = l%16) ----
  if (MODE == 1) {
    // gelu -> bf16, shuffled through LDS so global stores are b128-coalesced.
    constexpr int CP = 136;  // padded row stride (shorts); 128x136x2 = 34816B <= 36864B
    unsigned short (*Ct)[CP] = (unsigned short (*)[CP])(&As[0][0][0]);
    __syncthreads();   // all waves done reading K-loop LDS
#pragma unroll
    for (int mi = 0; mi < 2; ++mi) {
#pragma unroll
      for (int ni = 0; ni < 4; ++ni) {
        const int col = wn * 64 + ni * 16 + lrow;
        const float bv = bias[block_n + col];
#pragma unroll
        for (int i = 0; i < 8; ++i) {
          const int row = wm * 32 + mi * 16 + lhalf * 8 + i;
          Ct[row][col] = zb_f2bf(zb_gelu(acc[mi][ni][i] + bv));
        }
      }
    }
    __syncthreads();
    unsigned short* hid = (unsigned short*)Cv;
    const int orow = tid >> 1;              // 0..127
    const int ocol = (tid & 1) * 64;        // 0 or 64
    unsigned short* dst = hid + (size_t)(block_m + orow) * N + block_n + ocol;
#pragma unroll
    for (int j = 0; j < 8; ++j)
      *(uint4*)(dst + j * 8) = *(const uint4*)&Ct[orow][ocol + j * 8];
  } else {
    float* out = (float*)Cv;
#pragma unroll
    for (int mi = 0; mi < 2; ++mi) {
#pragma unroll
      for (int ni = 0; ni < 4; ++ni) {
        const int col = block_n + wn * 64 + ni * 16 + lrow;
        const float bv = bias[col];
        const int row0 = block_m + wm * 32 + mi * 16 + lhalf * 8;
        float* p = out + (size_t)row0 * N + col;
        float vals[8];
#pragma unroll
        for (int i = 0; i < 8; ++i) vals[i] = p[(size_t)i * N];   // batched loads
#pragma unroll
        for (int i = 0; i < 8; ++i)
          p[(size_t)i * N] = vals[i] + wgt * (acc[mi][ni][i] + bv);
      }
    }
  }
}

// ---------------- launch ----------------
extern "C" void kernel_launch(void* const* d_in, const int* in_sizes, int n_in,
                              void* d_out, int out_size, void* d_ws, size_t ws_size,
                              hipStream_t stream) {
  (void)in_sizes; (void)n_in; (void)out_size; (void)ws_size;
  const float* x   = (const float*)d_in[0];
  const float* rw1 = (const float*)d_in[1];
  const float* rb1 = (const float*)d_in[2];
  const float* rw2 = (const float*)d_in[3];
  const float* rb2 = (const float*)d_in[4];
  const float* zw1 = (const float*)d_in[5];
  const float* zb1 = (const float*)d_in[6];
  const float* zw2 = (const float*)d_in[7];
  const float* zb2 = (const float*)d_in[8];
  float* out = (float*)d_out;

  // workspace layout (256B aligned regions)
  char* ws = (char*)d_ws;
  const size_t SZ_X16 = (size_t)NTOK * D_MODEL * 2;     // 8 MB
  const size_t SZ_W   = (size_t)D_MODEL * D_FF * 2;     // 8 MB per expert matrix
  float*          w_be = (float*)(ws);
  unsigned short* x16  = (unsigned short*)(ws + 256);
  unsigned short* w1bT = (unsigned short*)(ws + 256 + SZ_X16);            // [F][D]
  unsigned short* w2bT = (unsigned short*)(ws + 256 + SZ_X16 + SZ_W);     // [D][F]
  unsigned short* hid  = (unsigned short*)(ws + 256 + SZ_X16 + 2 * SZ_W); // 32 MB [M][F]

  const int nX = NTOK * D_MODEL;          // 4M
  const size_t nW = (size_t)D_MODEL * D_FF;

  zb_f32_to_bf16<<<2048, 256, 0, stream>>>(x, x16, nX);
  zb_router<<<BATCH, 256, 0, stream>>>(x, rw1, rb1, rw2, rb2, w_be);
  zb_copy_f32<<<2048, 256, 0, stream>>>(x, out, nX / 4);

  for (int e = 0; e < N_EXPERTS; ++e) {
    // zw1[e]: [D_MODEL][D_FF] -> w1bT [D_FF][D_MODEL]
    zb_transpose_to_bf16<<<dim3(D_FF / 32, D_MODEL / 32), 256, 0, stream>>>(
        zw1 + e * nW, w1bT, D_MODEL, D_FF);
    // zw2[e]: [D_FF][D_MODEL] -> w2bT [D_MODEL][D_FF]
    zb_transpose_to_bf16<<<dim3(D_MODEL / 32, D_FF / 32), 256, 0, stream>>>(
        zw2 + e * nW, w2bT, D_FF, D_MODEL);
    // hid = gelu(x @ zw1[e] + zb1[e])            [4096 x 4096] bf16
    zb_moe_gemm<1><<<dim3(D_FF / 128, NTOK / 128), 256, 0, stream>>>(
        x16, w1bT, zb1 + (size_t)e * D_FF, (void*)hid, w_be,
        NTOK, D_FF, D_MODEL, e);
    // out += w_be[b,e] * (hid @ zw2[e] + zb2[e]) [4096 x 1024] f32
    zb_moe_gemm<2><<<dim3(D_MODEL / 128, NTOK / 128), 256, 0, stream>>>(
        hid, w2bT, zb2 + (size_t)e * D_MODEL, (void*)out, w_be,
        NTOK, D_MODEL, D_FF, e);
  }
}
